// LoRAMultiheadAttention_30185030156677
// MI455X (gfx1250) — matrix operand order repaired and checked by reading, not compiled, not run on hardware
//
#include <hip/hip_runtime.h>

// CDNA5 / gfx1250: wave32, WMMA f32_16x16x32_f16 (f16 mul, f32 accumulate).
typedef __attribute__((ext_vector_type(16))) _Float16 v16h;
typedef __attribute__((ext_vector_type(8)))  float    v8f;
typedef __attribute__((ext_vector_type(4)))  int      v4i;

union Frag16 { unsigned int u[8]; v16h h; };

#define AS1 __attribute__((address_space(1)))
#define AS3 __attribute__((address_space(3)))

#if defined(__has_builtin)
#if __has_builtin(__builtin_amdgcn_global_load_async_to_lds_b128) && \
    __has_builtin(__builtin_amdgcn_s_wait_asynccnt)
#define USE_ASYNC_LDS 1
#endif
#if __has_builtin(__builtin_amdgcn_update_dpp)
#define USE_DPP_RED 1
#endif
#if __has_builtin(__builtin_amdgcn_exp2f)
#define EXP2(x) __builtin_amdgcn_exp2f(x)
#else
#define EXP2(x) exp2f(x)
#endif
#endif

__device__ __forceinline__ void async_cp16(const _Float16* src, _Float16* dst) {
#ifdef USE_ASYNC_LDS
    __builtin_amdgcn_global_load_async_to_lds_b128(
        (AS1 v4i*)(src), (AS3 v4i*)(dst), 0, 0);
#else
    *(uint4*)dst = *(const uint4*)src;
#endif
}
__device__ __forceinline__ void async_fence() {
#ifdef USE_ASYNC_LDS
    __builtin_amdgcn_s_wait_asynccnt(0);
#endif
}

// 16-lane row reductions via DPP row_ror (VALU-only; no LDS traffic).
#ifdef USE_DPP_RED
template <int N>
__device__ __forceinline__ float row_ror(float x) {
    return __int_as_float(__builtin_amdgcn_update_dpp(
        0, __float_as_int(x), 0x120 | N, 0xf, 0xf, true));
}
__device__ __forceinline__ float redmax16(float x) {
    x = fmaxf(x, row_ror<1>(x)); x = fmaxf(x, row_ror<2>(x));
    x = fmaxf(x, row_ror<4>(x)); x = fmaxf(x, row_ror<8>(x));
    return x;
}
__device__ __forceinline__ float redsum16(float x) {
    x += row_ror<1>(x); x += row_ror<2>(x);
    x += row_ror<4>(x); x += row_ror<8>(x);
    return x;
}
#else
__device__ __forceinline__ float redmax16(float x) {
    for (int off = 8; off >= 1; off >>= 1) x = fmaxf(x, __shfl_xor(x, off, 32));
    return x;
}
__device__ __forceinline__ float redsum16(float x) {
    for (int off = 8; off >= 1; off >>= 1) x += __shfl_xor(x, off, 32);
    return x;
}
#endif

// Fragment layouts per CDNA5 ISA 7.12.2 (wave32):
//  A 16x32 f16 : lane m=lane&15, kq=lane>>4 ; dword v holds A[m][k0..k0+1],
//                k0 = (v<4?0:16) + kq*8 + (v&3)*2
//  B 32x16 f16 : lane n=lane&15, kq=lane>>4 ; dword v holds B[k0..k0+1][n],
//                k0 = (v<4?0:16) + kq*8 + (v&3)*2
//  C 16x16 f32 : lane n=lane&15, hl=lane>>4 ; vgpr r -> row r + 8*hl

// ---------------------------------------------------------------------------
// C = alpha * A(MxK) @ B(NxK)^T [+ bias] [+ C] ; f16 WMMA, f32 accumulate.
// Block tile 128x128, 256 threads = 8 waves (4x2), wave tile 32x64.
// Double-buffered LDS: loads for chunk k+1 issue before computing chunk k.
// ---------------------------------------------------------------------------
template <typename AT, typename OT, bool GUARD>
__global__ __launch_bounds__(256)
void wmma_gemm_xt(const AT* __restrict__ A, const float* __restrict__ B,
                  const float* __restrict__ bias, OT* __restrict__ C,
                  int M, int N, int K, float alpha, int accumulate)
{
    __shared__ alignas(16) _Float16 sA[2][128][32];
    __shared__ alignas(16) _Float16 sB[2][128][32];

    const int tid   = threadIdx.x;
    const int lane  = tid & 31;
    const int wave  = tid >> 5;
    const int waveM = wave >> 1;       // 0..3 : 32-row slice
    const int waveN = wave & 1;        // 0..1 : 64-col slice
    const int blockM = blockIdx.x * 128;
    const int blockN = blockIdx.y * 128;
    const int m0 = lane & 15, kq = lane >> 4, hl = kq;

    // staging: each thread owns a contiguous 16-elem run of one row
    const int srow = tid >> 1;          // 0..127
    const int scol = (tid & 1) * 16;    // 0 | 16

    const int nk = (K + 31) / 32;

    v8f acc[2][4];
    for (int mt = 0; mt < 2; ++mt)
        for (int nt = 0; nt < 4; ++nt)
            for (int r = 0; r < 8; ++r) acc[mt][nt][r] = 0.0f;

    _Float16 ta[16], tb[16];

    auto load_regs = [&](int k0) {
        const int gr = blockM + srow;
        if (!GUARD || (gr < M && (k0 + scol + 16) <= K)) {
            const AT* p = A + (size_t)gr * K + k0 + scol;
            if constexpr (sizeof(AT) == 2) {
                *(uint4*)&ta[0] = *(const uint4*)p;
                *(uint4*)&ta[8] = *(const uint4*)(p + 8);
            } else {
                const float4* q = (const float4*)p;
                for (int j = 0; j < 4; ++j) {
                    float4 t = q[j];
                    ta[4 * j + 0] = (_Float16)t.x; ta[4 * j + 1] = (_Float16)t.y;
                    ta[4 * j + 2] = (_Float16)t.z; ta[4 * j + 3] = (_Float16)t.w;
                }
            }
        } else {
            for (int j = 0; j < 16; ++j) {
                int c = k0 + scol + j;
                float v = (gr < M && c < K) ? (float)A[(size_t)gr * K + c] : 0.0f;
                ta[j] = (_Float16)v;
            }
        }
        const int gn = blockN + srow;
        if (!GUARD || (gn < N && (k0 + scol + 16) <= K)) {
            const float4* q = (const float4*)(B + (size_t)gn * K + k0 + scol);
            for (int j = 0; j < 4; ++j) {
                float4 t = q[j];
                tb[4 * j + 0] = (_Float16)t.x; tb[4 * j + 1] = (_Float16)t.y;
                tb[4 * j + 2] = (_Float16)t.z; tb[4 * j + 3] = (_Float16)t.w;
            }
        } else {
            for (int j = 0; j < 16; ++j) {
                int c = k0 + scol + j;
                float v = (gn < N && c < K) ? B[(size_t)gn * K + c] : 0.0f;
                tb[j] = (_Float16)v;
            }
        }
    };
    auto store_lds = [&](int b) {
        *(uint4*)&sA[b][srow][scol]     = *(uint4*)&ta[0];
        *(uint4*)&sA[b][srow][scol + 8] = *(uint4*)&ta[8];
        *(uint4*)&sB[b][srow][scol]     = *(uint4*)&tb[0];
        *(uint4*)&sB[b][srow][scol + 8] = *(uint4*)&tb[8];
    };
    auto compute = [&](int b) {
        Frag16 af[2], bf[4];
        for (int mt = 0; mt < 2; ++mt)
            for (int v = 0; v < 8; ++v) {
                int kk = ((v < 4) ? 0 : 16) + kq * 8 + (v & 3) * 2;
                af[mt].u[v] = *(const unsigned int*)&sA[b][waveM * 32 + mt * 16 + m0][kk];
            }
        for (int nt = 0; nt < 4; ++nt)
            for (int v = 0; v < 8; ++v) {
                int kk = ((v < 4) ? 0 : 16) + kq * 8 + (v & 3) * 2;
                bf[nt].u[v] = *(const unsigned int*)&sB[b][waveN * 64 + nt * 16 + m0][kk];
            }
        for (int mt = 0; mt < 2; ++mt)
            for (int nt = 0; nt < 4; ++nt)
                acc[mt][nt] = __builtin_amdgcn_wmma_f32_16x16x32_f16(
                    false, af[mt].h, false, bf[nt].h, (short)0, acc[mt][nt], false, false);
    };

    load_regs(0);
    store_lds(0);
    __syncthreads();
    for (int kc = 0; kc < nk; ++kc) {
        const int cur = kc & 1;
        if (kc + 1 < nk) {
            load_regs((kc + 1) * 32);
            if (!GUARD && (kc + 2) < nk) {
                __builtin_prefetch(A + (size_t)(blockM + srow) * K + (kc + 2) * 32 + scol, 0, 3);
                __builtin_prefetch(B + (size_t)(blockN + srow) * K + (kc + 2) * 32 + scol, 0, 3);
            }
        }
        compute(cur);
        if (kc + 1 < nk) store_lds(cur ^ 1);
        __syncthreads();
    }

    for (int mt = 0; mt < 2; ++mt)
        for (int nt = 0; nt < 4; ++nt)
            for (int r = 0; r < 8; ++r) {
                int row = blockM + waveM * 32 + mt * 16 + r + 8 * hl;
                int col = blockN + waveN * 64 + nt * 16 + m0;
                if (!GUARD || (row < M && col < N)) {
                    float v = acc[mt][nt][r] * alpha;
                    if (bias) v += bias[col];
                    size_t idx = (size_t)row * N + col;
                    if (accumulate) v += (float)C[idx];
                    C[idx] = (OT)v;
                }
            }
}

// ---------------------------------------------------------------------------
// Flash attention. qkv: (L, N, 3E) f16. One WG = (head, 64 query rows),
// 128 threads = 4 waves, each wave owns a 16-row q tile. Key blocks of 64,
// double-buffered in LDS (async K copies + register-staged transposed V).
// Softmax runs in the log2 domain: 0.125*log2(e) is folded into Q once.
// All staging loops have compile-time trip counts so vst[] stays in VGPRs.
// ---------------------------------------------------------------------------
#define L_SEQ 2048
#define NB    2
#define EMB   1024
#define HEADS 16
#define HD    64
#define KB    64
#define NKB   (L_SEQ / KB)

__global__ __launch_bounds__(128)
void flash_attn(const _Float16* __restrict__ qkv, _Float16* __restrict__ out)
{
    __shared__ alignas(16) _Float16 sQ[64][64];        // [qrow][d]
    __shared__ alignas(16) _Float16 sK[2][KB][64];     // [key][d]
    __shared__ alignas(16) _Float16 sVt[2][64][KB];    // [d][key] (transposed)
    __shared__ alignas(16) _Float16 sP[4][16][KB];     // per-wave P relayout

    const int head = blockIdx.x;              // 0..31 (= n*H + h)
    const int qb   = blockIdx.y;              // 0..31
    const int n    = head / HEADS;
    const int h    = head % HEADS;
    const int tid  = threadIdx.x;
    const int lane = tid & 31;
    const int wave = tid >> 5;                // 0..3
    const int m0 = lane & 15, kq = lane >> 4, hl = kq;

    const int    rowStride = NB * 3 * EMB;    // elems between consecutive l
    const size_t baseQ = (size_t)n * 3 * EMB + (size_t)h * HD;
    const size_t baseK = baseQ + EMB;
    const size_t baseV = baseQ + 2 * EMB;

    uint4 vst[4];  // register-staged V tile (4 x 16B segments per thread)

    auto load_k_async = [&](int kb_, int b) {
        for (int j = 0; j < 4; ++j) {              // 512 segs / 128 thr = 4
            const int i = tid + j * 128;
            const int key = i >> 3, seg = i & 7;
            async_cp16(qkv + (size_t)(kb_ * KB + key) * rowStride + baseK + seg * 8,
                       &sK[b][key][seg * 8]);
        }
    };
    auto load_v = [&](int kb_) {
        for (int j = 0; j < 4; ++j) {
            const int i = tid + j * 128;
            const int key = i >> 3, seg = i & 7;
            vst[j] = *(const uint4*)(qkv + (size_t)(kb_ * KB + key) * rowStride +
                                     baseV + seg * 8);
        }
    };
    auto store_v = [&](int b) {
        for (int j = 0; j < 4; ++j) {
            const int i = tid + j * 128;
            const int key = i >> 3, seg = i & 7;
            const _Float16* th = (const _Float16*)&vst[j];
            for (int jj = 0; jj < 8; ++jj) sVt[b][seg * 8 + jj][key] = th[jj];
        }
    };

    // ---- prologue: Q + block 0 ----
    for (int j = 0; j < 4; ++j) {
        const int i = tid + j * 128;
        const int r = i >> 3, seg = i & 7;
        async_cp16(qkv + (size_t)(qb * 64 + r) * rowStride + baseQ + seg * 8,
                   &sQ[r][seg * 8]);
    }
    load_k_async(0, 0);
    load_v(0);
    async_fence();
    store_v(0);
    __syncthreads();

    // preload this wave's Q A-fragments, folding in 0.125 * log2(e)
    Frag16 aq[2];
    for (int kc = 0; kc < 2; ++kc) {
        for (int v = 0; v < 8; ++v) {
            int d = kc * 32 + ((v < 4) ? 0 : 16) + kq * 8 + (v & 3) * 2;
            aq[kc].u[v] = *(const unsigned int*)&sQ[wave * 16 + m0][d];
        }
        aq[kc].h = aq[kc].h * (_Float16)0.18033688f;  // 0.125 * log2(e)
    }

    float mrow[8], lrow[8];
    v8f   accO[4];
    for (int r = 0; r < 8; ++r) { mrow[r] = -1e30f; lrow[r] = 0.0f; }
    for (int t = 0; t < 4; ++t)
        for (int r = 0; r < 8; ++r) accO[t][r] = 0.0f;

    for (int kb = 0; kb < NKB; ++kb) {
        const int cur = kb & 1;
        if (kb + 1 < NKB) {           // issue next block's loads before compute
            load_k_async(kb + 1, cur ^ 1);
            load_v(kb + 1);
        }

        // S = Qs @ K^T : 4 key tiles of 16, K-dim = 64 (2 chunks)
        v8f s[4];
        for (int nt = 0; nt < 4; ++nt) {
            Frag16 bk[2];
            for (int kc = 0; kc < 2; ++kc)
                for (int v = 0; v < 8; ++v) {
                    int d = kc * 32 + ((v < 4) ? 0 : 16) + kq * 8 + (v & 3) * 2;
                    bk[kc].u[v] = *(const unsigned int*)&sK[cur][nt * 16 + m0][d];
                }
            v8f z;
            for (int r = 0; r < 8; ++r) z[r] = 0.0f;
            z = __builtin_amdgcn_wmma_f32_16x16x32_f16(false, aq[0].h, false, bk[0].h, (short)0, z, false, false);
            z = __builtin_amdgcn_wmma_f32_16x16x32_f16(false, aq[1].h, false, bk[1].h, (short)0, z, false, false);
            s[nt] = z;
        }

        // online softmax in log2 domain; DPP rotate reductions over 16 lanes
        float pscale[8];
        for (int r = 0; r < 8; ++r) {
            float mx = fmaxf(fmaxf(s[0][r], s[1][r]), fmaxf(s[2][r], s[3][r]));
            mx = redmax16(mx);
            float mnew = fmaxf(mrow[r], mx);
            pscale[r]  = EXP2(mrow[r] - mnew);
            mrow[r]    = mnew;
            float sum = 0.0f;
            for (int nt = 0; nt < 4; ++nt) {
                float p = EXP2(s[nt][r] - mnew);
                s[nt][r] = p;
                sum += p;
            }
            sum = redsum16(sum);
            lrow[r] = lrow[r] * pscale[r] + sum;
        }
        for (int t = 0; t < 4; ++t)
            for (int r = 0; r < 8; ++r) accO[t][r] *= pscale[r];

        // relayout P (C layout -> row-major LDS -> A fragments); per-wave
        // buffer, DS ops from one wave are in order -> no barrier needed.
        for (int nt = 0; nt < 4; ++nt)
            for (int r = 0; r < 8; ++r)
                sP[wave][r + 8 * hl][nt * 16 + m0] = (_Float16)s[nt][r];
        Frag16 ap[2];
        for (int kc = 0; kc < 2; ++kc)
            for (int v = 0; v < 8; ++v) {
                int k = kc * 32 + ((v < 4) ? 0 : 16) + kq * 8 + (v & 3) * 2;
                ap[kc].u[v] = *(const unsigned int*)&sP[wave][m0][k];
            }

        // O += P @ V : 4 d-tiles, K-dim = 64 keys (pairs contiguous in sVt)
        for (int t = 0; t < 4; ++t) {
            Frag16 bv[2];
            for (int kc = 0; kc < 2; ++kc)
                for (int v = 0; v < 8; ++v) {
                    int k = kc * 32 + ((v < 4) ? 0 : 16) + kq * 8 + (v & 3) * 2;
                    bv[kc].u[v] = *(const unsigned int*)&sVt[cur][t * 16 + m0][k];
                }
            accO[t] = __builtin_amdgcn_wmma_f32_16x16x32_f16(
                false, ap[0].h, false, bv[0].h, (short)0, accO[t], false, false);
            accO[t] = __builtin_amdgcn_wmma_f32_16x16x32_f16(
                false, ap[1].h, false, bv[1].h, (short)0, accO[t], false, false);
        }

        if (kb + 1 < NKB) store_v(cur ^ 1);
        async_fence();
        __syncthreads();
    }

    // normalize and write O in (L, N, E) layout, f16
    for (int t = 0; t < 4; ++t)
        for (int r = 0; r < 8; ++r) {
            int row = qb * 64 + wave * 16 + r + 8 * hl;   // = l
            int d   = t * 16 + m0;
            float v = accO[t][r] / lrow[r];
            out[(size_t)row * (NB * EMB) + (size_t)n * EMB + h * HD + d] = (_Float16)v;
        }
}

// ---------------------------------------------------------------------------
extern "C" void kernel_launch(void* const* d_in, const int* in_sizes, int n_in,
                              void* d_out, int out_size, void* d_ws, size_t ws_size,
                              hipStream_t stream)
{
    const float* query  = (const float*)d_in[0];
    const float* in_w   = (const float*)d_in[3];   // (3E, E)
    const float* in_b   = (const float*)d_in[4];   // (3E,)
    const float* out_w  = (const float*)d_in[5];   // (E, E)
    const float* out_b  = (const float*)d_in[6];   // (E,)
    const float* lora_A = (const float*)d_in[7];   // (16, E)
    const float* lora_B = (const float*)d_in[8];   // (E, 16)

    const int M = L_SEQ * NB;        // 4096 rows (l*N + n)
    _Float16* qkv_h  = (_Float16*)d_ws;                      // 4096 x 3072 f16
    _Float16* attn_h = qkv_h  + (size_t)M * 3 * EMB;         // 4096 x 1024 f16
    _Float16* t_h    = attn_h + (size_t)M * EMB;             // 4096 x 16   f16
    float*    outp   = (float*)d_out;

    dim3 blk(256);

    // 1) qkv = X @ Win^T + bias  (f32 in, f16 out, full tiles)
    wmma_gemm_xt<float, _Float16, false><<<dim3(M / 128, (3 * EMB) / 128), blk, 0, stream>>>(
        query, in_w, in_b, qkv_h, M, 3 * EMB, EMB, 1.0f, 0);

    // 2) flash attention over 32 heads x 32 query blocks
    flash_attn<<<dim3(NB * HEADS, L_SEQ / 64), dim3(128), 0, stream>>>(qkv_h, attn_h);

    // 3) T = attn @ loraA^T  (N = 16, guarded)
    wmma_gemm_xt<_Float16, _Float16, true><<<dim3(M / 128, 1), blk, 0, stream>>>(
        attn_h, lora_A, nullptr, t_h, M, 16, EMB, 1.0f, 0);

    // 4) out = attn @ Wout^T + bias  (full tiles)
    wmma_gemm_xt<_Float16, float, false><<<dim3(M / 128, EMB / 128), blk, 0, stream>>>(
        attn_h, out_w, out_b, outp, M, EMB, EMB, 1.0f, 0);

    // 5) out += (T @ loraB^T) * LORA_SCALING  (K = 16, guarded, accumulate)
    wmma_gemm_xt<_Float16, float, true><<<dim3(M / 128, EMB / 128), blk, 0, stream>>>(
        t_h, lora_B, nullptr, outp, M, EMB, 16, 2.0f, 1);
}
